// bayes_linear_LR_2L_38637525794845
// MI455X (gfx1250) — matrix operand order repaired and checked by reading, not compiled, not run on hardware
//
#include <hip/hip_runtime.h>
#include <hip/hip_bf16.h>

typedef __attribute__((ext_vector_type(16))) _Float16 v16h;
typedef __attribute__((ext_vector_type(8)))  float    v8f;

union AFrag { v16h v; uint4 q[2]; };

__device__ __forceinline__ float softplusf(float x) {
  return (x > 20.f) ? x : log1pf(expf(x));
}
// order-preserving float<->uint encoding (atomicMax on uint == max on float)
__device__ __forceinline__ unsigned fenc(float f) {
  unsigned u = __float_as_uint(f);
  return (u & 0x80000000u) ? ~u : (u | 0x80000000u);
}
__device__ __forceinline__ float fdec(unsigned u) {
  unsigned b = (u & 0x80000000u) ? (u & 0x7fffffffu) : ~u;
  return __uint_as_float(b);
}

// ---- CDNA5 async global->LDS (ASYNCcnt-tracked DMA, no VGPR round trip) ----
__device__ __forceinline__ void async_ld_b128(const void* gptr, unsigned lds_off) {
  asm volatile("global_load_async_to_lds_b128 %0, %1, off"
               :: "v"(lds_off), "v"((unsigned long long)(uintptr_t)gptr)
               : "memory");
}
__device__ __forceinline__ void wait_async0() {
#if __has_builtin(__builtin_amdgcn_s_wait_asynccnt)
  __builtin_amdgcn_s_wait_asynccnt(0);
#else
  asm volatile("s_wait_asynccnt 0x0" ::: "memory");
#endif
}
__device__ __forceinline__ unsigned lds_off32(const void* p) {
  return (unsigned)(uintptr_t)p;  // low 32 bits of generic LDS pointer = LDS byte offset
}

// ---------------- small utility kernels ----------------
__global__ void k_zero(unsigned* smax, float* accs, int zero_accs) {
  int t = threadIdx.x;
  if (t < 10) smax[t] = 0u;
  if (zero_accs && t < 2) accs[t] = 0.f;   // accs[0]=kld, accs[1]=lip
}

__global__ void k_cvt(const float* __restrict__ x, _Float16* __restrict__ xh, size_t n) {
  size_t i = (size_t)blockIdx.x * blockDim.x + threadIdx.x;
  if (i < n) xh[i] = (_Float16)x[i];
}

__global__ void k_copy(const float* __restrict__ s, float* __restrict__ d, int n) {
  int i = blockIdx.x * 256 + threadIdx.x;
  if (i < n) d[i] = s[i];
}

__global__ void k_wout(const float* __restrict__ accs, float* __restrict__ out, size_t ofs) {
  if (threadIdx.x == 0) { out[ofs] = accs[0]; out[ofs + 1] = accs[1]; }
}

// ---------------- weight prep: f16 transpose + sigma + KLD partials ----------------
__global__ __launch_bounds__(256) void k_prep_w(
    const float* __restrict__ Wmu, const float* __restrict__ Wp,
    _Float16* __restrict__ WmuT, _Float16* __restrict__ S2T,
    float* __restrict__ sigW, float* __restrict__ kpart, int ni, int no) {
  __shared__ _Float16 tW[32 * 33];
  __shared__ _Float16 tS[32 * 33];
  __shared__ float red[256];
  int j0 = blockIdx.x * 32, i0 = blockIdx.y * 32;
  int c = threadIdx.x & 31, r4 = threadIdx.x >> 5;
  float kl = 0.f;
#pragma unroll
  for (int q = 0; q < 4; ++q) {
    int r = r4 + 8 * q;
    size_t idx = (size_t)(i0 + r) * no + (j0 + c);
    float mu = Wmu[idx], p = Wp[idx];
    float sg = 1e-6f + softplusf(p);
    sigW[idx] = sg;
    kl += 2.f * logf(0.1f / sg) - 1.f + 100.f * sg * sg + 100.f * mu * mu;
    tW[r * 33 + c] = (_Float16)mu;
    tS[r * 33 + c] = (_Float16)(sg * sg);
  }
  red[threadIdx.x] = kl;
  __syncthreads();
#pragma unroll
  for (int q = 0; q < 4; ++q) {
    int r = r4 + 8 * q;
    size_t odx = (size_t)(j0 + r) * ni + (i0 + c);
    WmuT[odx] = tW[c * 33 + r];
    S2T[odx]  = tS[c * 33 + r];
  }
  for (int s = 128; s > 0; s >>= 1) {
    if (threadIdx.x < s) red[threadIdx.x] += red[threadIdx.x + s];
    __syncthreads();
  }
  if (threadIdx.x == 0) kpart[blockIdx.y * gridDim.x + blockIdx.x] = red[0];
}

// ---------------- bias prep + bias KLD ----------------
__global__ __launch_bounds__(256) void k_bias(
    const float* __restrict__ bmu, const float* __restrict__ bp,
    const float* __restrict__ epsb, float* __restrict__ bias,
    float* __restrict__ kldb, int no) {
  __shared__ float red[256];
  float kl = 0.f;
  for (int n = threadIdx.x; n < no; n += 256) {
    float sg = 1e-6f + softplusf(bp[n]);
    bias[n] = bmu[n] + sg * epsb[n];
    kl += 2.f * logf(0.1f / sg) - 1.f + 100.f * sg * sg + 100.f * bmu[n] * bmu[n];
  }
  red[threadIdx.x] = kl;
  __syncthreads();
  for (int s = 128; s > 0; s >>= 1) {
    if (threadIdx.x < s) red[threadIdx.x] += red[threadIdx.x + s];
    __syncthreads();
  }
  if (threadIdx.x == 0) *kldb = red[0];
}

// ---------------- fused dual WMMA GEMM: mu = x@W, var = x^2@sigma^2 ----------------
// A: Xh [M][K] f16 (x). B: WT/S2T [N][K] f16 (pre-transposed).
// Staging: async global->LDS (ASYNCcnt) with explicit double buffering.
// out = mu + sqrt(var)*eps + bias, optional relu; writes f32 y or f16 next-layer input.
__global__ __launch_bounds__(256) void k_gemm(
    const _Float16* __restrict__ Xh, const _Float16* __restrict__ WT,
    const _Float16* __restrict__ S2T, const float* __restrict__ epsw,
    const float* __restrict__ bias, float* __restrict__ Yout,
    _Float16* __restrict__ Hout, int M, int N, int K, int do_relu) {
  __shared__ __align__(16) _Float16 Al[2][128 * 40];
  __shared__ __align__(16) _Float16 Bwl[2][64 * 40];
  __shared__ __align__(16) _Float16 Bsl[2][64 * 40];
  const int tid = threadIdx.x;
  const int lane = tid & 31, wid = tid >> 5;
  const int wm = wid >> 1, wn = wid & 1;
  const int bm0 = blockIdx.y * 128, bn0 = blockIdx.x * 64;

  // per-thread staging coordinates (b128 granules)
  const int rA0 = tid >> 2;                 // rows 0..63 of A tile
  const int rA1 = rA0 + 64;                 // rows 64..127
  const int cA  = (tid & 3) * 8;            // half-column within 32-wide K slab
  const _Float16* gA0 = Xh  + (size_t)(bm0 + rA0) * K + cA;
  const _Float16* gA1 = Xh  + (size_t)(bm0 + rA1) * K + cA;
  const _Float16* gBw = WT  + (size_t)(bn0 + rA0) * K + cA;
  const _Float16* gBs = S2T + (size_t)(bn0 + rA0) * K + cA;
  const unsigned lA0 = lds_off32(&Al[0][rA0 * 40 + cA]);
  const unsigned lA1 = lds_off32(&Al[0][rA1 * 40 + cA]);
  const unsigned lBw = lds_off32(&Bwl[0][rA0 * 40 + cA]);
  const unsigned lBs = lds_off32(&Bsl[0][rA0 * 40 + cA]);
  const unsigned dA = (unsigned)(sizeof(_Float16) * 128 * 40);
  const unsigned dB = (unsigned)(sizeof(_Float16) * 64 * 40);

  v8f mu[2][2], va[2][2];
  const v8f vzero = {0.f, 0.f, 0.f, 0.f, 0.f, 0.f, 0.f, 0.f};
#pragma unroll
  for (int mi = 0; mi < 2; ++mi)
#pragma unroll
    for (int nj = 0; nj < 2; ++nj) { mu[mi][nj] = vzero; va[mi][nj] = vzero; }

  const int nk = K >> 5;
  // prologue: stage slab 0 into buffer 0
  async_ld_b128(gA0, lA0);
  async_ld_b128(gA1, lA1);
  async_ld_b128(gBw, lBw);
  async_ld_b128(gBs, lBs);
  wait_async0();
  __syncthreads();

  for (int it = 0; it < nk; ++it) {
    const int buf = it & 1;
    if (it + 1 < nk) {  // prefetch next slab into the other buffer (async DMA)
      const int k1 = (it + 1) << 5;
      const unsigned bofs = (it + 1) & 1;
      async_ld_b128(gA0 + k1, lA0 + bofs * dA);
      async_ld_b128(gA1 + k1, lA1 + bofs * dA);
      async_ld_b128(gBw + k1, lBw + bofs * dB);
      async_ld_b128(gBs + k1, lBs + bofs * dB);
    }

    AFrag a[2], a2[2], bw[2], bs[2];
#pragma unroll
    for (int mi = 0; mi < 2; ++mi) {
      int rl = wm * 32 + mi * 16 + (lane & 15);
      int ko = (lane >> 4) * 8;  // lanes 0-15: K{0..7,16..23}; 16-31: K{8..15,24..31}
      a[mi].q[0] = *(const uint4*)(&Al[buf][rl * 40 + ko]);
      a[mi].q[1] = *(const uint4*)(&Al[buf][rl * 40 + ko + 16]);
      a2[mi].v = a[mi].v * a[mi].v;  // x^2 fragment in-register
    }
#pragma unroll
    for (int nj = 0; nj < 2; ++nj) {
      int nl = wn * 32 + nj * 16 + (lane & 15);
      int ko = (lane >> 4) * 8;  // lanes 0-15: K{0..7,16..23}; 16-31: K{8..15,24..31}
      bw[nj].q[0] = *(const uint4*)(&Bwl[buf][nl * 40 + ko]);
      bw[nj].q[1] = *(const uint4*)(&Bwl[buf][nl * 40 + ko + 16]);
      bs[nj].q[0] = *(const uint4*)(&Bsl[buf][nl * 40 + ko]);
      bs[nj].q[1] = *(const uint4*)(&Bsl[buf][nl * 40 + ko + 16]);
    }
#pragma unroll
    for (int mi = 0; mi < 2; ++mi)
#pragma unroll
      for (int nj = 0; nj < 2; ++nj) {
        mu[mi][nj] = __builtin_amdgcn_wmma_f32_16x16x32_f16(
            false, a[mi].v, false, bw[nj].v, (short)0, mu[mi][nj], false, false);
        va[mi][nj] = __builtin_amdgcn_wmma_f32_16x16x32_f16(
            false, a2[mi].v, false, bs[nj].v, (short)0, va[mi][nj], false, false);
      }

    if (it + 1 < nk) {  // drain this wave's async DMAs, then rendezvous
      wait_async0();
      __syncthreads();
    }
  }

  // epilogue: C layout lanes0-15 N=lane M=v; lanes16-31 N=lane-16 M=v+8
  int cr = lane & 15, rofs = (lane >> 4) * 8;
#pragma unroll
  for (int mi = 0; mi < 2; ++mi)
#pragma unroll
    for (int nj = 0; nj < 2; ++nj)
#pragma unroll
      for (int e = 0; e < 8; ++e) {
        int row = bm0 + wm * 32 + mi * 16 + rofs + e;
        int col = bn0 + wn * 32 + nj * 16 + cr;
        size_t idx = (size_t)row * N + col;
        float v = mu[mi][nj][e] +
                  sqrtf(fmaxf(va[mi][nj][e], 0.f)) * epsw[idx] + bias[col];
        if (do_relu) v = fmaxf(v, 0.f);
        if (Yout) Yout[idx] = v;
        else Hout[idx] = (_Float16)v;
      }
}

// ---------------- row/col squared norms of sigma ----------------
__global__ __launch_bounds__(256) void k_rowsq(const float* __restrict__ sigW,
                                               float* __restrict__ rowsq, int ni, int no) {
  int row = blockIdx.x * 8 + (threadIdx.x >> 5);
  int lane = threadIdx.x & 31;
  float acc = 0.f;
  for (int j = lane; j < no; j += 32) { float s = sigW[(size_t)row * no + j]; acc += s * s; }
  for (int o = 16; o; o >>= 1) acc += __shfl_xor(acc, o, 32);
  if (lane == 0) rowsq[row] = acc;
}

__global__ __launch_bounds__(256) void k_colsq(const float* __restrict__ sigW,
                                               float* __restrict__ colsq, int ni, int no) {
  int col = blockIdx.x * 256 + threadIdx.x;
  float acc = 0.f;
  for (int i = 0; i < ni; ++i) { float s = sigW[(size_t)i * no + col]; acc += s * s; }
  colsq[col] = acc;
}

// ---------------- max over sim[s]*sigma, per s (atomicMax, order-independent) -------
__global__ __launch_bounds__(256) void k_simmax(const float* __restrict__ sim,
                                                const float* __restrict__ sigW,
                                                unsigned* __restrict__ smax, size_t nel) {
  __shared__ float red[256];
  int s = blockIdx.y;
  size_t base = (size_t)blockIdx.x * 2048;
  const float* sp = sim + (size_t)s * nel;
  float m = -3.4e38f;
#pragma unroll
  for (int k = 0; k < 8; ++k) {
    size_t e = base + (size_t)k * 256 + threadIdx.x;
    m = fmaxf(m, sp[e] * sigW[e]);
  }
  red[threadIdx.x] = m;
  __syncthreads();
  for (int t = 128; t > 0; t >>= 1) {
    if (threadIdx.x < t) red[threadIdx.x] = fmaxf(red[threadIdx.x], red[threadIdx.x + t]);
    __syncthreads();
  }
  if (threadIdx.x == 0) atomicMax(&smax[s], fenc(red[0]));
}

// ---------------- GEMV kernels for power iteration (unnormalized) ----------------
__global__ __launch_bounds__(256) void k_mv_t(const float* __restrict__ W,
                                              const float* __restrict__ u,
                                              float* __restrict__ v, int ni, int no) {
  int j = blockIdx.x * 256 + threadIdx.x;
  float acc = 0.f;
  for (int i = 0; i < ni; ++i) acc += W[(size_t)i * no + j] * u[i];
  v[j] = acc;
}

__global__ __launch_bounds__(256) void k_mv(const float* __restrict__ W,
                                            const float* __restrict__ v,
                                            float* __restrict__ u, int ni, int no) {
  int row = blockIdx.x * 8 + (threadIdx.x >> 5);
  int lane = threadIdx.x & 31;
  float acc = 0.f;
  for (int j = lane; j < no; j += 32) acc += W[(size_t)row * no + j] * v[j];
  for (int o = 16; o; o >>= 1) acc += __shfl_xor(acc, o, 32);
  if (lane == 0) u[row] = acc;
}

// ---------------- per-layer finalize: res + tmp + sigma, KLD, accumulate ----------
__global__ __launch_bounds__(256) void k_fin(
    const float* __restrict__ rowsq, const float* __restrict__ colsq,
    const unsigned* __restrict__ smax, const float* __restrict__ ubuf,
    const float* __restrict__ vbuf, const float* __restrict__ kpart,
    const float* __restrict__ kldb, float* __restrict__ accs,
    int ni, int no, int npart) {
  __shared__ float red[256];
  int t = threadIdx.x;
  float m = -3.4e38f;
  for (int i = t; i < ni; i += 256) m = fmaxf(m, rowsq[i]);
  red[t] = m; __syncthreads();
  for (int s = 128; s > 0; s >>= 1) { if (t < s) red[t] = fmaxf(red[t], red[t + s]); __syncthreads(); }
  float rmax = sqrtf(red[0]); __syncthreads();

  m = -3.4e38f;
  for (int j = t; j < no; j += 256) m = fmaxf(m, colsq[j]);
  red[t] = m; __syncthreads();
  for (int s = 128; s > 0; s >>= 1) { if (t < s) red[t] = fmaxf(red[t], red[t + s]); __syncthreads(); }
  float cmax = sqrtf(red[0]); __syncthreads();

  float a = 0.f;
  for (int i = t; i < ni; i += 256) { float x = ubuf[i]; a += x * x; }
  red[t] = a; __syncthreads();
  for (int s = 128; s > 0; s >>= 1) { if (t < s) red[t] += red[t + s]; __syncthreads(); }
  float su = red[0]; __syncthreads();

  a = 0.f;
  for (int j = t; j < no; j += 256) { float x = vbuf[j]; a += x * x; }
  red[t] = a; __syncthreads();
  for (int s = 128; s > 0; s >>= 1) { if (t < s) red[t] += red[t + s]; __syncthreads(); }
  float sv = red[0]; __syncthreads();

  a = 0.f;
  for (int p = t; p < npart; p += 256) a += kpart[p];
  red[t] = a; __syncthreads();
  for (int s = 128; s > 0; s >>= 1) { if (t < s) red[t] += red[t + s]; __syncthreads(); }

  if (t == 0) {
    float kw = red[0];
    float tmp = 0.f;
    for (int s = 0; s < 10; ++s) tmp += fdec(smax[s]);
    tmp *= 0.1f;
    float sigma = sqrtf(su / sv);  // = ||W v_hat|| for unnormalized chain
    float lip = rmax + cmax + tmp + sigma;
    accs[1] += lip * lip;
    accs[0] += 0.5f * (kw + kldb[0]);
  }
}

// ---------------- host orchestration ----------------
extern "C" void kernel_launch(void* const* d_in, const int* in_sizes, int n_in,
                              void* d_out, int out_size, void* d_ws, size_t ws_size,
                              hipStream_t stream) {
  (void)in_sizes; (void)n_in; (void)ws_size;
  const int B = 8192;
  const int NI[3] = {1024, 2048, 2048};
  const int NO[3] = {2048, 2048, 1024};
  const float* x = (const float*)d_in[0];
  const float *Wmu[3], *Wp[3], *bmu[3], *bp[3], *epsw[3], *epsb[3], *uin[3], *sim[3];
  for (int l = 0; l < 3; ++l) {
    int b = 1 + l * 8;
    Wmu[l] = (const float*)d_in[b + 0];  Wp[l]   = (const float*)d_in[b + 1];
    bmu[l] = (const float*)d_in[b + 2];  bp[l]   = (const float*)d_in[b + 3];
    epsw[l] = (const float*)d_in[b + 4]; epsb[l] = (const float*)d_in[b + 5];
    uin[l] = (const float*)d_in[b + 6];  sim[l]  = (const float*)d_in[b + 7];
  }
  char* ws = (char*)d_ws;
  size_t off = 0;
  auto alloc = [&](size_t bytes) -> char* {
    char* p = ws + off;
    off = (off + bytes + 255) & ~(size_t)255;
    return p;
  };
  _Float16* WmuT = (_Float16*)alloc((size_t)2048 * 2048 * 2);
  _Float16* S2T  = (_Float16*)alloc((size_t)2048 * 2048 * 2);
  float*    sigW = (float*)alloc((size_t)2048 * 2048 * 4);
  _Float16* xhA  = (_Float16*)alloc((size_t)B * 2048 * 2);
  _Float16* xhB  = (_Float16*)alloc((size_t)B * 2048 * 2);
  float*    bias = (float*)alloc(2048 * 4);
  float*    ubuf = (float*)alloc(2048 * 4);
  float*    vbuf = (float*)alloc(2048 * 4);
  float*    rowsq = (float*)alloc(2048 * 4);
  float*    colsq = (float*)alloc(2048 * 4);
  unsigned* smax = (unsigned*)alloc(10 * 4);
  float*    kpart = (float*)alloc(4096 * 4);
  float*    kldb = (float*)alloc(4);
  float*    accs = (float*)alloc(2 * 4);

  { size_t n = (size_t)B * 1024;
    k_cvt<<<dim3((unsigned)((n + 255) / 256)), 256, 0, stream>>>(x, xhA, n); }

  float* yout = (float*)d_out;
  for (int l = 0; l < 3; ++l) {
    int ni = NI[l], no = NO[l];
    k_zero<<<1, 64, 0, stream>>>(smax, accs, l == 0 ? 1 : 0);
    dim3 pg(no / 32, ni / 32);
    k_prep_w<<<pg, 256, 0, stream>>>(Wmu[l], Wp[l], WmuT, S2T, sigW, kpart, ni, no);
    k_bias<<<1, 256, 0, stream>>>(bmu[l], bp[l], epsb[l], bias, kldb, no);

    const _Float16* xin = (l == 1) ? xhB : xhA;
    _Float16* hout = (l == 0) ? xhB : ((l == 1) ? xhA : nullptr);
    float* yo = (l == 2) ? yout : nullptr;
    dim3 gg(no / 64, B / 128);
    k_gemm<<<gg, 256, 0, stream>>>(xin, WmuT, S2T, epsw[l], bias, yo, hout,
                                   B, no, ni, (l < 2) ? 1 : 0);

    k_rowsq<<<ni / 8, 256, 0, stream>>>(sigW, rowsq, ni, no);
    k_colsq<<<no / 256, 256, 0, stream>>>(sigW, colsq, ni, no);
    size_t nel = (size_t)ni * no;
    dim3 sg((unsigned)(nel / 2048), 10);
    k_simmax<<<sg, 256, 0, stream>>>(sim[l], sigW, smax, nel);

    k_copy<<<(ni + 255) / 256, 256, 0, stream>>>(uin[l], ubuf, ni);
    for (int it = 0; it < 10; ++it) {
      k_mv_t<<<no / 256, 256, 0, stream>>>(Wmu[l], ubuf, vbuf, ni, no);
      k_mv<<<ni / 8, 256, 0, stream>>>(Wmu[l], vbuf, ubuf, ni, no);
    }
    int npart = (no / 32) * (ni / 32);
    k_fin<<<1, 256, 0, stream>>>(rowsq, colsq, smax, ubuf, vbuf, kpart, kldb,
                                 accs, ni, no, npart);
  }
  k_wout<<<1, 32, 0, stream>>>(accs, yout, (size_t)out_size - 2);
}
